// DecoderBlock_38628935860430
// MI455X (gfx1250) — matrix operand order repaired and checked by reading, not compiled, not run on hardware
//
#include <hip/hip_runtime.h>

// ---------------- problem constants ----------------
constexpr int TT   = 2048;   // sequence length (B=1)
constexpr int DIMC = 2048;
constexpr int NQH  = 16;
constexpr int NKVH = 4;
constexpr int HDC  = 128;
constexpr int NEC  = 8;
constexpr int EHC  = 4096;
constexpr int PMAX = 4224;   // 2048*2 assignments + 8*16 alignment pad (divisible by 64)
constexpr float NEGF = -1000000000.0f;
constexpr float LN_ROPE_BASE = 15.424948470398375f; // ln(5e6)

typedef __attribute__((ext_vector_type(16))) __bf16 v16bf;
typedef __attribute__((ext_vector_type(8)))  float  v8f;
typedef __attribute__((ext_vector_type(4)))  unsigned int v4u;

union Frag { v16bf v; v4u q[2]; };

__device__ inline v8f v8zero() {
  v8f v;
  #pragma unroll
  for (int i = 0; i < 8; ++i) v[i] = 0.0f;
  return v;
}

__device__ inline v8f wmma_bf16(const v16bf a, const v16bf b, const v8f c) {
  return __builtin_amdgcn_wmma_f32_16x16x32_bf16(false, a, false, b, (short)0, c, false, false);
}

// load A fragment: row-major A[M,K] bf16; lane row = m0 + (lane&15)
// elements i<8 -> k = k0 + half*8 + i ; i>=8 -> k = k0 + half*8 + 16 + (i-8)
__device__ inline v16bf load_a_frag(const __bf16* rowptr, int k0, int half) {
  Frag f;
  const __bf16* p = rowptr + k0 + half * 8;
  f.q[0] = *(const v4u*)(p);
  f.q[1] = *(const v4u*)(p + 16);
  return f.v;
}
// load B fragment: Bt[N,K] row-major bf16; lane col n = n0 + (lane&15)
// elements i<8 -> k = k0 + half*8 + i ; i>=8 -> k = k0 + half*8 + 16 + (i-8)
__device__ inline v16bf load_b_frag(const __bf16* rowptr, int k0, int half) {
  Frag f;
  const __bf16* p = rowptr + k0 + half * 8;
  f.q[0] = *(const v4u*)(p);
  f.q[1] = *(const v4u*)(p + 16);
  return f.v;
}

// ---------------- transpose + convert f32 -> bf16 (batched) ----------------
// in: [batch, R, C] f32  ->  out: [batch, C, R] bf16
__global__ __launch_bounds__(256)
void convT_kernel(const float* __restrict__ in, __bf16* __restrict__ out,
                  int R, int C, long long total) {
  long long i = (long long)blockIdx.x * 256 + threadIdx.x;
  if (i >= total) return;
  long long rc = (long long)R * C;
  long long b = i / rc, rem = i - b * rc;
  int r = (int)(rem / C), c = (int)(rem - (long long)r * C);
  out[b * rc + (long long)c * R + r] = (__bf16)in[i];
}

// ---------------- RMSNorm over DIMC ----------------
__global__ __launch_bounds__(256)
void rmsnorm_kernel(const float* __restrict__ x, const float* __restrict__ w,
                    __bf16* __restrict__ outB, float* __restrict__ outF) {
  int row = blockIdx.x;
  const float* xr = x + (size_t)row * DIMC;
  float ss = 0.f;
  for (int c = threadIdx.x; c < DIMC; c += 256) { float v = xr[c]; ss += v * v; }
  #pragma unroll
  for (int o = 1; o < 32; o <<= 1) ss += __shfl_xor(ss, o, 32);
  __shared__ float red[8];
  int wid = threadIdx.x >> 5, lane = threadIdx.x & 31;
  if (lane == 0) red[wid] = ss;
  __syncthreads();
  float tot = 0.f;
  #pragma unroll
  for (int i = 0; i < 8; ++i) tot += red[i];
  float sc = rsqrtf(tot / (float)DIMC + 1e-6f);
  for (int c = threadIdx.x; c < DIMC; c += 256) {
    float v = xr[c] * sc * w[c];
    outB[(size_t)row * DIMC + c] = (__bf16)v;
    if (outF) outF[(size_t)row * DIMC + c] = v;
  }
}

// ---------------- generic GEMM: C[M,N] = A[M,K](bf16) x Bt[N,K](bf16) ----------------
// block = 128 threads = 4 waves (M stacked); wave tile 16x64
// k-loop unrolled by 2 with ping-pong register buffers: compute on buffer A
// while buffer B's loads are in flight, then swap (no register copies).
// Requires K % 64 == 0 (true for all uses here).
__global__ __launch_bounds__(128)
void gemm_bf16_kernel(const __bf16* __restrict__ A, const __bf16* __restrict__ Bt,
                      int M, int N, int K,
                      const float* __restrict__ resid,
                      float* __restrict__ outF, __bf16* __restrict__ outB) {
  int wave = threadIdx.x >> 5, lane = threadIdx.x & 31;
  int half = lane >> 4, l16 = lane & 15;
  int m0 = blockIdx.y * 64 + wave * 16;
  int n0 = blockIdx.x * 64;
  if (m0 >= M) return;
  v8f acc[4];
  #pragma unroll
  for (int t = 0; t < 4; ++t) acc[t] = v8zero();
  const __bf16* Arow = A + (size_t)(m0 + l16) * K;
  const __bf16* Brow[4];
  #pragma unroll
  for (int t = 0; t < 4; ++t) Brow[t] = Bt + (size_t)(n0 + t * 16 + l16) * K;

  v16bf afA = load_a_frag(Arow, 0, half);
  v16bf bfA[4];
  #pragma unroll
  for (int t = 0; t < 4; ++t) bfA[t] = load_b_frag(Brow[t], 0, half);

  for (int k0 = 0; k0 < K; k0 += 64) {
    int k1 = k0 + 32;                       // always < K (K % 64 == 0)
    v16bf afB = load_a_frag(Arow, k1, half);
    v16bf bfB[4];
    #pragma unroll
    for (int t = 0; t < 4; ++t) bfB[t] = load_b_frag(Brow[t], k1, half);
    #pragma unroll
    for (int t = 0; t < 4; ++t) acc[t] = wmma_bf16(afA, bfA[t], acc[t]);

    int k2 = (k0 + 64 < K) ? k0 + 64 : 0;   // wraps on last iter (dead prefetch)
    afA = load_a_frag(Arow, k2, half);
    #pragma unroll
    for (int t = 0; t < 4; ++t) bfA[t] = load_b_frag(Brow[t], k2, half);
    #pragma unroll
    for (int t = 0; t < 4; ++t) acc[t] = wmma_bf16(afB, bfB[t], acc[t]);
  }
  #pragma unroll
  for (int t = 0; t < 4; ++t) {
    #pragma unroll
    for (int r = 0; r < 8; ++r) {
      int row = m0 + r + half * 8;
      int col = n0 + t * 16 + l16;
      float v = acc[t][r];
      if (resid) v += resid[(size_t)row * N + col];
      if (outF) outF[(size_t)row * N + col] = v;
      if (outB) outB[(size_t)row * N + col] = (__bf16)v;
    }
  }
}

// ---------------- per-head RMSNorm + RoPE, write bf16 head-major ----------------
// grid (TT, NQH+NKVH), 128 threads (one per dim element)
__global__ __launch_bounds__(128)
void qkrope_kernel(const float* __restrict__ qf, const float* __restrict__ kf,
                   const float* __restrict__ qnw, const float* __restrict__ knw,
                   __bf16* __restrict__ Qb, __bf16* __restrict__ Kb) {
  int t = blockIdx.x, hh = blockIdx.y, d = threadIdx.x;
  const float* src; const float* w; __bf16* dst;
  if (hh < NQH) {
    src = qf + (size_t)t * (NQH * HDC) + hh * HDC; w = qnw;
    dst = Qb + ((size_t)hh * TT + t) * HDC;
  } else {
    int kv = hh - NQH;
    src = kf + (size_t)t * (NKVH * HDC) + kv * HDC; w = knw;
    dst = Kb + ((size_t)kv * TT + t) * HDC;
  }
  float v = src[d];
  float ss = v * v;
  #pragma unroll
  for (int o = 1; o < 32; o <<= 1) ss += __shfl_xor(ss, o, 32);
  __shared__ float red[4];
  int wid = threadIdx.x >> 5, lane = threadIdx.x & 31;
  if (lane == 0) red[wid] = ss;
  __syncthreads();
  float tot = red[0] + red[1] + red[2] + red[3];
  float sc = rsqrtf(tot / (float)HDC + 1e-6f);
  float xn = v * sc * w[d];
  int dp = d ^ 64;
  float xp = src[dp] * sc * w[dp];
  float inv = expf(-((float)(2 * (d & 63)) / (float)HDC) * LN_ROPE_BASE);
  float ang = (float)t * inv;
  float cs = cosf(ang), sn = sinf(ang);
  float rot = (d < 64) ? -xp : xp;
  dst[d] = (__bf16)(xn * cs + rot * sn);
}

// ---------------- V transpose: vf[T, NKVH*HDC] f32 -> Vt[NKVH][HDC][T] bf16 ----------------
__global__ __launch_bounds__(256)
void vtrans_kernel(const float* __restrict__ vf, __bf16* __restrict__ Vt) {
  int i = blockIdx.x * 256 + threadIdx.x;
  if (i >= TT * NKVH * HDC) return;
  int t = i / (NKVH * HDC), rem = i % (NKVH * HDC);
  int kv = rem / HDC, d = rem % HDC;
  Vt[((size_t)kv * HDC + d) * TT + t] = (__bf16)vf[i];
}

// ---------------- flash attention: 1 wave per (16-query tile, head) ----------------
__global__ __launch_bounds__(32)
void flash_attn_kernel(const __bf16* __restrict__ Qb, const __bf16* __restrict__ Kb,
                       const __bf16* __restrict__ Vt, __bf16* __restrict__ ctx) {
  __shared__ __bf16 Pls[16 * 32];
  int lane = threadIdx.x & 31;
  int half = lane >> 4, l16 = lane & 15;
  int qt = blockIdx.x, h = blockIdx.y;
  int kh = h / (NQH / NKVH);
  int q0 = qt * 16;
  const float sc = 0.08838834764831845f; // 1/sqrt(128)

  Frag qfr[4];
  const __bf16* qrow = Qb + ((size_t)h * TT + q0 + l16) * HDC;
  #pragma unroll
  for (int c = 0; c < 4; ++c) qfr[c].v = load_a_frag(qrow, c * 32, half);

  v8f O[8];
  #pragma unroll
  for (int t = 0; t < 8; ++t) O[t] = v8zero();
  float m[8], lsum[8];
  #pragma unroll
  for (int r = 0; r < 8; ++r) { m[r] = -1e30f; lsum[r] = 0.f; }

  int iters = (q0 + 16 + 31) / 32;
  for (int it = 0; it < iters; ++it) {
    int j0 = it * 32;
    v8f s0 = v8zero(), s1 = v8zero();
    #pragma unroll
    for (int c = 0; c < 4; ++c) {
      const __bf16* kr0 = Kb + ((size_t)kh * TT + j0 + l16) * HDC;
      const __bf16* kr1 = kr0 + (size_t)16 * HDC;
      s0 = wmma_bf16(qfr[c].v, load_b_frag(kr0, c * 32, half), s0);
      s1 = wmma_bf16(qfr[c].v, load_b_frag(kr1, c * 32, half), s1);
    }
    #pragma unroll
    for (int r = 0; r < 8; ++r) {
      int q = q0 + r + half * 8;
      float a0 = s0[r] * sc + ((j0 + l16) > q ? NEGF : 0.f);
      float a1 = s1[r] * sc + ((j0 + 16 + l16) > q ? NEGF : 0.f);
      float mx = fmaxf(a0, a1);
      #pragma unroll
      for (int o = 1; o < 16; o <<= 1) mx = fmaxf(mx, __shfl_xor(mx, o, 32));
      float mn = fmaxf(m[r], mx);
      float alpha = __expf(m[r] - mn);
      m[r] = mn;
      float p0 = __expf(a0 - mn), p1 = __expf(a1 - mn);
      float ps = p0 + p1;
      #pragma unroll
      for (int o = 1; o < 16; o <<= 1) ps += __shfl_xor(ps, o, 32);
      lsum[r] = lsum[r] * alpha + ps;
      s0[r] = p0; s1[r] = p1;
      #pragma unroll
      for (int t = 0; t < 8; ++t) O[t][r] *= alpha;
    }
    __syncthreads();
    #pragma unroll
    for (int r = 0; r < 8; ++r) {
      Pls[(r + half * 8) * 32 + l16]      = (__bf16)s0[r];
      Pls[(r + half * 8) * 32 + 16 + l16] = (__bf16)s1[r];
    }
    __syncthreads();
    Frag pf;
    const __bf16* pr = &Pls[l16 * 32 + half * 8];
    pf.q[0] = *(const v4u*)(pr);
    pf.q[1] = *(const v4u*)(pr + 16);
    #pragma unroll
    for (int t = 0; t < 8; ++t) {
      const __bf16* vr = Vt + ((size_t)kh * HDC + t * 16 + l16) * TT + j0;
      O[t] = wmma_bf16(pf.v, load_b_frag(vr, 0, half), O[t]);
    }
    __syncthreads();
  }
  #pragma unroll
  for (int t = 0; t < 8; ++t) {
    #pragma unroll
    for (int r = 0; r < 8; ++r) {
      int row = q0 + r + half * 8;
      ctx[(size_t)row * (NQH * HDC) + h * HDC + t * 16 + l16] =
          (__bf16)(O[t][r] / lsum[r]);
    }
  }
}

// ---------------- gate GEMV + softmax + top-2, 1 wave per token ----------------
__global__ __launch_bounds__(32)
void gate_topk_kernel(const float* __restrict__ hf, const float* __restrict__ wgate,
                      float* __restrict__ topv, int* __restrict__ topi) {
  int n = blockIdx.x, lane = threadIdx.x;
  float acc[NEC];
  #pragma unroll
  for (int e = 0; e < NEC; ++e) acc[e] = 0.f;
  const float* hr = hf + (size_t)n * DIMC;
  for (int d = lane; d < DIMC; d += 32) {
    float x = hr[d];
    const float* wr = wgate + (size_t)d * NEC;
    #pragma unroll
    for (int e = 0; e < NEC; ++e) acc[e] += x * wr[e];
  }
  #pragma unroll
  for (int e = 0; e < NEC; ++e) {
    #pragma unroll
    for (int o = 1; o < 32; o <<= 1) acc[e] += __shfl_xor(acc[e], o, 32);
  }
  if (lane == 0) {
    float mx = acc[0];
    #pragma unroll
    for (int e = 1; e < NEC; ++e) mx = fmaxf(mx, acc[e]);
    float p[NEC], s = 0.f;
    #pragma unroll
    for (int e = 0; e < NEC; ++e) { p[e] = __expf(acc[e] - mx); s += p[e]; }
    #pragma unroll
    for (int e = 0; e < NEC; ++e) p[e] /= s;
    int i0 = 0; float v0 = p[0];
    #pragma unroll
    for (int e = 1; e < NEC; ++e) if (p[e] > v0) { v0 = p[e]; i0 = e; }
    int i1 = -1; float v1 = -1.f;
    #pragma unroll
    for (int e = 0; e < NEC; ++e) if (e != i0 && p[e] > v1) { v1 = p[e]; i1 = e; }
    topv[n * 2] = v0; topv[n * 2 + 1] = v1;
    topi[n * 2] = i0; topi[n * 2 + 1] = i1;
  }
}

// ---------------- MoE routing helpers ----------------
__global__ void fill_i32_kernel(int* p, int val, int n) {
  int i = blockIdx.x * 256 + threadIdx.x;
  if (i < n) p[i] = val;
}
__global__ void moe_count_kernel(const int* __restrict__ topi, int* cnt) {
  int n = blockIdx.x * 256 + threadIdx.x;
  if (n < TT) { atomicAdd(&cnt[topi[n * 2]], 1); atomicAdd(&cnt[topi[n * 2 + 1]], 1); }
}
__global__ void moe_base_kernel(const int* __restrict__ cnt, int* base, int* cnt2) {
  if (threadIdx.x == 0 && blockIdx.x == 0) {
    int o = 0;
    for (int e = 0; e < NEC; ++e) { base[e] = o; o += (cnt[e] + 15) & ~15; cnt2[e] = 0; }
    base[NEC] = o;
  }
}
__global__ void moe_assign_kernel(const int* __restrict__ topi, const int* __restrict__ base,
                                  int* cnt2, int* __restrict__ rows, int* __restrict__ tok) {
  int n = blockIdx.x * 256 + threadIdx.x;
  if (n >= TT) return;
  for (int j = 0; j < 2; ++j) {
    int e = topi[n * 2 + j];
    int pos = atomicAdd(&cnt2[e], 1);
    int r = base[e] + pos;
    rows[n * 2 + j] = r;
    tok[r] = n;
  }
}
__global__ __launch_bounds__(256)
void gather_kernel(const __bf16* __restrict__ hb, const int* __restrict__ tok,
                   __bf16* __restrict__ Xg) {
  int r = blockIdx.x;
  int t = tok[r];
  for (int c = threadIdx.x; c < DIMC; c += 256)
    Xg[(size_t)r * DIMC + c] = (t >= 0) ? hb[(size_t)t * DIMC + c] : (__bf16)0.0f;
}

__device__ inline int find_expert(const int* base, int m0) {
  int e = NEC - 1;
  #pragma unroll
  for (int i = 0; i < NEC; ++i) {
    if (m0 < base[i + 1]) { e = i; break; }
  }
  return e;
}

// ---------------- MoE GEMM1: act = silu(Xg x wgT[e]) * (Xg x wiT[e]) ----------------
__global__ __launch_bounds__(128)
void moe_gemm1_kernel(const __bf16* __restrict__ Xg, const __bf16* __restrict__ wiT,
                      const __bf16* __restrict__ wgT, const int* __restrict__ base,
                      __bf16* __restrict__ act) {
  int wave = threadIdx.x >> 5, lane = threadIdx.x & 31;
  int half = lane >> 4, l16 = lane & 15;
  int m0 = blockIdx.y * 64 + wave * 16;
  int n0 = blockIdx.x * 64;
  int e = find_expert(base, m0);
  const __bf16* B1row[4];
  const __bf16* B2row[4];
  #pragma unroll
  for (int t = 0; t < 4; ++t) {
    size_t boff = (size_t)(n0 + t * 16 + l16) * DIMC;
    B1row[t] = wiT + (size_t)e * EHC * DIMC + boff;
    B2row[t] = wgT + (size_t)e * EHC * DIMC + boff;
  }
  v8f up[4], ga[4];
  #pragma unroll
  for (int t = 0; t < 4; ++t) { up[t] = v8zero(); ga[t] = v8zero(); }
  const __bf16* Arow = Xg + (size_t)(m0 + l16) * DIMC;

  v16bf afA = load_a_frag(Arow, 0, half);
  v16bf b1A[4], b2A[4];
  #pragma unroll
  for (int t = 0; t < 4; ++t) {
    b1A[t] = load_b_frag(B1row[t], 0, half);
    b2A[t] = load_b_frag(B2row[t], 0, half);
  }
  for (int k0 = 0; k0 < DIMC; k0 += 64) {
    int k1 = k0 + 32;
    v16bf afB = load_a_frag(Arow, k1, half);
    v16bf b1B[4], b2B[4];
    #pragma unroll
    for (int t = 0; t < 4; ++t) {
      b1B[t] = load_b_frag(B1row[t], k1, half);
      b2B[t] = load_b_frag(B2row[t], k1, half);
    }
    #pragma unroll
    for (int t = 0; t < 4; ++t) {
      up[t] = wmma_bf16(afA, b1A[t], up[t]);
      ga[t] = wmma_bf16(afA, b2A[t], ga[t]);
    }
    int k2 = (k0 + 64 < DIMC) ? k0 + 64 : 0;
    afA = load_a_frag(Arow, k2, half);
    #pragma unroll
    for (int t = 0; t < 4; ++t) {
      b1A[t] = load_b_frag(B1row[t], k2, half);
      b2A[t] = load_b_frag(B2row[t], k2, half);
    }
    #pragma unroll
    for (int t = 0; t < 4; ++t) {
      up[t] = wmma_bf16(afB, b1B[t], up[t]);
      ga[t] = wmma_bf16(afB, b2B[t], ga[t]);
    }
  }
  #pragma unroll
  for (int t = 0; t < 4; ++t) {
    #pragma unroll
    for (int r = 0; r < 8; ++r) {
      int row = m0 + r + half * 8;
      int col = n0 + t * 16 + l16;
      float g = ga[t][r];
      float a = (g / (1.f + __expf(-g))) * up[t][r];
      act[(size_t)row * EHC + col] = (__bf16)a;
    }
  }
}

// ---------------- MoE GEMM2: y = act x woT[e] ----------------
__global__ __launch_bounds__(128)
void moe_gemm2_kernel(const __bf16* __restrict__ act, const __bf16* __restrict__ woT,
                      const int* __restrict__ base, float* __restrict__ y) {
  int wave = threadIdx.x >> 5, lane = threadIdx.x & 31;
  int half = lane >> 4, l16 = lane & 15;
  int m0 = blockIdx.y * 64 + wave * 16;
  int n0 = blockIdx.x * 64;
  int e = find_expert(base, m0);
  const __bf16* Brow[4];
  #pragma unroll
  for (int t = 0; t < 4; ++t)
    Brow[t] = woT + (size_t)e * DIMC * EHC + (size_t)(n0 + t * 16 + l16) * EHC;
  v8f acc[4];
  #pragma unroll
  for (int t = 0; t < 4; ++t) acc[t] = v8zero();
  const __bf16* Arow = act + (size_t)(m0 + l16) * EHC;

  v16bf afA = load_a_frag(Arow, 0, half);
  v16bf bfA[4];
  #pragma unroll
  for (int t = 0; t < 4; ++t) bfA[t] = load_b_frag(Brow[t], 0, half);

  for (int k0 = 0; k0 < EHC; k0 += 64) {
    int k1 = k0 + 32;
    v16bf afB = load_a_frag(Arow, k1, half);
    v16bf bfB[4];
    #pragma unroll
    for (int t = 0; t < 4; ++t) bfB[t] = load_b_frag(Brow[t], k1, half);
    #pragma unroll
    for (int t = 0; t < 4; ++t) acc[t] = wmma_bf16(afA, bfA[t], acc[t]);

    int k2 = (k0 + 64 < EHC) ? k0 + 64 : 0;
    afA = load_a_frag(Arow, k2, half);
    #pragma unroll
    for (int t = 0; t < 4; ++t) bfA[t] = load_b_frag(Brow[t], k2, half);
    #pragma unroll
    for (int t = 0; t < 4; ++t) acc[t] = wmma_bf16(afB, bfB[t], acc[t]);
  }
  #pragma unroll
  for (int t = 0; t < 4; ++t) {
    #pragma unroll
    for (int r = 0; r < 8; ++r) {
      int row = m0 + r + half * 8;
      int col = n0 + t * 16 + l16;
      y[(size_t)row * DIMC + col] = acc[t][r];
    }
  }
}

// ---------------- final combine: out = x1 + w0*y[r0] + w1*y[r1] ----------------
__global__ __launch_bounds__(256)
void combine_kernel(const float* __restrict__ x1, const float* __restrict__ y,
                    const float* __restrict__ topv, const int* __restrict__ rows,
                    float* __restrict__ out) {
  int n = blockIdx.x;
  int r0 = rows[n * 2], r1 = rows[n * 2 + 1];
  float w0 = topv[n * 2], w1 = topv[n * 2 + 1];
  for (int c = threadIdx.x; c < DIMC; c += 256)
    out[(size_t)n * DIMC + c] = x1[(size_t)n * DIMC + c]
        + w0 * y[(size_t)r0 * DIMC + c] + w1 * y[(size_t)r1 * DIMC + c];
}

// =====================================================================
extern "C" void kernel_launch(void* const* d_in, const int* in_sizes, int n_in,
                              void* d_out, int out_size, void* d_ws, size_t ws_size,
                              hipStream_t stream) {
  const float* x     = (const float*)d_in[0];
  const float* Wq    = (const float*)d_in[2];
  const float* Wk    = (const float*)d_in[3];
  const float* Wv    = (const float*)d_in[4];
  const float* Wo    = (const float*)d_in[5];
  const float* qnw   = (const float*)d_in[6];
  const float* knw   = (const float*)d_in[7];
  const float* anw   = (const float*)d_in[8];
  const float* fnw   = (const float*)d_in[9];
  const float* wgate = (const float*)d_in[10];
  const float* wi    = (const float*)d_in[11];
  const float* wg    = (const float*)d_in[12];
  const float* wo    = (const float*)d_in[13];
  float* out = (float*)d_out;

  char* wsb = (char*)d_ws;
  size_t off = 0;
  auto alloc = [&](size_t bytes) -> void* {
    off = (off + 255) & ~(size_t)255;
    void* p = wsb + off; off += bytes; return p;
  };
  __bf16* WqT  = (__bf16*)alloc((size_t)DIMC * (NQH * HDC) * 2);
  __bf16* WkT  = (__bf16*)alloc((size_t)DIMC * (NKVH * HDC) * 2);
  __bf16* WvT  = (__bf16*)alloc((size_t)DIMC * (NKVH * HDC) * 2);
  __bf16* WoT  = (__bf16*)alloc((size_t)(NQH * HDC) * DIMC * 2);
  __bf16* wiT  = (__bf16*)alloc((size_t)NEC * DIMC * EHC * 2);
  __bf16* wgT  = (__bf16*)alloc((size_t)NEC * DIMC * EHC * 2);
  __bf16* woT  = (__bf16*)alloc((size_t)NEC * EHC * DIMC * 2);
  __bf16* ab   = (__bf16*)alloc((size_t)TT * DIMC * 2);
  float*  qf   = (float*) alloc((size_t)TT * NQH * HDC * 4);
  float*  kf   = (float*) alloc((size_t)TT * NKVH * HDC * 4);
  float*  vf   = (float*) alloc((size_t)TT * NKVH * HDC * 4);
  __bf16* Qb   = (__bf16*)alloc((size_t)NQH * TT * HDC * 2);
  __bf16* Kb   = (__bf16*)alloc((size_t)NKVH * TT * HDC * 2);
  __bf16* Vt   = (__bf16*)alloc((size_t)NKVH * HDC * TT * 2);
  __bf16* ctxb = (__bf16*)alloc((size_t)TT * NQH * HDC * 2);
  float*  x1   = (float*) alloc((size_t)TT * DIMC * 4);
  float*  hf   = (float*) alloc((size_t)TT * DIMC * 4);
  __bf16* hb   = (__bf16*)alloc((size_t)TT * DIMC * 2);
  float*  topv = (float*) alloc((size_t)TT * 2 * 4);
  int*    topi = (int*)   alloc((size_t)TT * 2 * 4);
  int*    rows = (int*)   alloc((size_t)TT * 2 * 4);
  int*    cnt  = (int*)   alloc(16 * 4);
  int*    cnt2 = (int*)   alloc(16 * 4);
  int*    base = (int*)   alloc(16 * 4);
  int*    tok  = (int*)   alloc((size_t)PMAX * 4);
  __bf16* Xg   = (__bf16*)alloc((size_t)PMAX * DIMC * 2);
  __bf16* actb = (__bf16*)alloc((size_t)PMAX * EHC * 2);
  float*  yb   = (float*) alloc((size_t)PMAX * DIMC * 4);
  (void)ws_size; (void)in_sizes; (void)n_in; (void)out_size;

  auto launch_convT = [&](const float* in, __bf16* o, int R, int C, int batch) {
    long long total = (long long)batch * R * C;
    int blocks = (int)((total + 255) / 256);
    convT_kernel<<<blocks, 256, 0, stream>>>(in, o, R, C, total);
  };
  // weight conversions (every call; deterministic)
  launch_convT(Wq, WqT, DIMC, NQH * HDC, 1);
  launch_convT(Wk, WkT, DIMC, NKVH * HDC, 1);
  launch_convT(Wv, WvT, DIMC, NKVH * HDC, 1);
  launch_convT(Wo, WoT, NQH * HDC, DIMC, 1);
  launch_convT(wi, wiT, DIMC, EHC, NEC);
  launch_convT(wg, wgT, DIMC, EHC, NEC);
  launch_convT(wo, woT, EHC, DIMC, NEC);

  // attention input norm
  rmsnorm_kernel<<<TT, 256, 0, stream>>>(x, anw, ab, nullptr);

  // QKV projections
  gemm_bf16_kernel<<<dim3((NQH * HDC) / 64, TT / 64), 128, 0, stream>>>(
      ab, WqT, TT, NQH * HDC, DIMC, nullptr, qf, nullptr);
  gemm_bf16_kernel<<<dim3((NKVH * HDC) / 64, TT / 64), 128, 0, stream>>>(
      ab, WkT, TT, NKVH * HDC, DIMC, nullptr, kf, nullptr);
  gemm_bf16_kernel<<<dim3((NKVH * HDC) / 64, TT / 64), 128, 0, stream>>>(
      ab, WvT, TT, NKVH * HDC, DIMC, nullptr, vf, nullptr);

  // per-head norm + rope; V transpose
  qkrope_kernel<<<dim3(TT, NQH + NKVH), 128, 0, stream>>>(qf, kf, qnw, knw, Qb, Kb);
  vtrans_kernel<<<(TT * NKVH * HDC + 255) / 256, 256, 0, stream>>>(vf, Vt);

  // attention
  flash_attn_kernel<<<dim3(TT / 16, NQH), 32, 0, stream>>>(Qb, Kb, Vt, ctxb);

  // output projection + residual
  gemm_bf16_kernel<<<dim3(DIMC / 64, TT / 64), 128, 0, stream>>>(
      ctxb, WoT, TT, DIMC, NQH * HDC, x, x1, nullptr);

  // ffn norm
  rmsnorm_kernel<<<TT, 256, 0, stream>>>(x1, fnw, hb, hf);

  // gating + routing
  gate_topk_kernel<<<TT, 32, 0, stream>>>(hf, wgate, topv, topi);
  fill_i32_kernel<<<1, 256, 0, stream>>>(cnt, 0, NEC);
  moe_count_kernel<<<(TT + 255) / 256, 256, 0, stream>>>(topi, cnt);
  moe_base_kernel<<<1, 32, 0, stream>>>(cnt, base, cnt2);
  fill_i32_kernel<<<(PMAX + 255) / 256, 256, 0, stream>>>(tok, -1, PMAX);
  moe_assign_kernel<<<(TT + 255) / 256, 256, 0, stream>>>(topi, base, cnt2, rows, tok);
  gather_kernel<<<PMAX, 256, 0, stream>>>(hb, tok, Xg);

  // expert GEMMs
  moe_gemm1_kernel<<<dim3(EHC / 64, PMAX / 64), 128, 0, stream>>>(Xg, wiT, wgT, base, actb);
  moe_gemm2_kernel<<<dim3(DIMC / 64, PMAX / 64), 128, 0, stream>>>(actb, woT, base, yb);

  // combine
  combine_kernel<<<TT, 256, 0, stream>>>(x1, yb, topv, rows, out);
}
